// MetaLayer_3315714752553
// MI455X (gfx1250) — matrix operand order repaired and checked by reading, not compiled, not run on hardware
//
#include <hip/hip_runtime.h>

// MI455X (gfx1250) fused meta-layer.
// Compute-bound (~430 GFLOP vs ~140MB HBM traffic): all 13 large GEMMs go
// through v_wmma_f32_16x16x32_bf16 (bf16 inputs, f32 accumulate).
// wave32; 256-thread blocks = 8 waves; 128x128x32 block tile; LDS-staged.

#define D_ 512
#define B_ 16
#define T_ 1024

typedef __bf16 bf16;
typedef __attribute__((ext_vector_type(16))) __bf16 v16bf;
typedef __attribute__((ext_vector_type(8)))  float  v8f;

union FragBF { v16bf v; unsigned int u[8]; };

// ---------------------------------------------------------------- GEMM ----
// C[M,N] (f32) = (transA ? A^T : A)(bf16) @ B(bf16) [+ bias] [+ C if accum]
// A row-major (lda), or K x M when transA. B row-major K x N (ldb).
// Batched over blockIdx.z with element strides sA/sB/sC/sBias (0 = shared).
#define BM 128
#define BN 128
#define BK 32

__global__ __launch_bounds__(256)
void gemm_bf16(const bf16* __restrict__ A0, long long sA, int lda, int transA,
               const bf16* __restrict__ B0, long long sB, int ldb,
               const float* __restrict__ bias0, long long sBias,
               float* __restrict__ C0, long long sC, int ldc,
               int M, int N, int K, int accum)
{
    __shared__ bf16 lA[BM][BK + 8];   // 128 x 40 bf16
    __shared__ bf16 lB[BK][BN + 8];   // 32 x 136 bf16

    const int tid  = threadIdx.x;
    const int lane = tid & 31;
    const int wid  = tid >> 5;
    const int wm   = wid & 3;          // 4 row-waves * 32 rows
    const int wn   = wid >> 2;         // 2 col-waves * 64 cols

    const int m0 = blockIdx.y * BM;
    const int n0 = blockIdx.x * BN;
    const long long zb = blockIdx.z;

    const bf16*  A    = A0 + zb * sA;
    const bf16*  Bm   = B0 + zb * sB;
    float*       C    = C0 + zb * sC;
    const float* bias = bias0 ? (bias0 + zb * sBias) : (const float*)0;

    v8f acc[2][4];
    #pragma unroll
    for (int i = 0; i < 2; ++i)
      #pragma unroll
      for (int j = 0; j < 4; ++j)
        #pragma unroll
        for (int e = 0; e < 8; ++e) acc[i][j][e] = 0.0f;

    for (int k0 = 0; k0 < K; k0 += BK) {
        if (!transA) {
            #pragma unroll
            for (int i = 0; i < 2; ++i) {
                int c = tid + i * 256;                 // 512 chunks of 8 bf16
                int r = c >> 2;
                int kk = (c & 3) * 8;
                int4 v = *(const int4*)(A + (long long)(m0 + r) * lda + k0 + kk);
                *(int4*)&lA[r][kk] = v;
            }
        } else {
            #pragma unroll
            for (int i = 0; i < 2; ++i) {
                int c = tid + i * 256;
                int kr = c >> 4;                       // 0..31
                int mc = (c & 15) * 8;
                union { int4 i4; bf16 h[8]; } u;
                u.i4 = *(const int4*)(A + (long long)(k0 + kr) * lda + m0 + mc);
                #pragma unroll
                for (int e = 0; e < 8; ++e) lA[mc + e][kr] = u.h[e];
            }
        }
        #pragma unroll
        for (int i = 0; i < 2; ++i) {
            int c = tid + i * 256;
            int kr = c >> 4;
            int nc = (c & 15) * 8;
            int4 v = *(const int4*)(Bm + (long long)(k0 + kr) * ldb + n0 + nc);
            *(int4*)&lB[kr][nc] = v;
        }
        __syncthreads();

        // A frag (16x32 bf16, ISA layout): lane m=lane&15, half g=lane>>4,
        // VGPR v<4: K=g*8+2v+{0,1}; v>=4: K=16+g*8+2(v-4)+{0,1}.
        FragBF aF[2], bF[4];
        const int mrow = lane & 15;
        const int g    = lane >> 4;
        #pragma unroll
        for (int mi = 0; mi < 2; ++mi) {
            int m = wm * 32 + mi * 16 + mrow;
            #pragma unroll
            for (int v = 0; v < 4; ++v) {
                aF[mi].u[v]     = *(const unsigned int*)&lA[m][g * 8 + v * 2];
                aF[mi].u[4 + v] = *(const unsigned int*)&lA[m][16 + g * 8 + v * 2];
            }
        }
        // B frag (32x16): N=lane&15, K in same order as A frag.
        #pragma unroll
        for (int ni = 0; ni < 4; ++ni) {
            int n = wn * 64 + ni * 16 + (lane & 15);
            #pragma unroll
            for (int j = 0; j < 8; ++j) {
                bF[ni].v[j]     = lB[g * 8 + j][n];
                bF[ni].v[8 + j] = lB[16 + g * 8 + j][n];
            }
        }
        #pragma unroll
        for (int mi = 0; mi < 2; ++mi)
          #pragma unroll
          for (int ni = 0; ni < 4; ++ni)
            acc[mi][ni] = __builtin_amdgcn_wmma_f32_16x16x32_bf16(
                false, aF[mi].v, false, bF[ni].v, (short)0, acc[mi][ni],
                false, false);
        __syncthreads();
    }

    // D layout: VGPR r, lanes 0-15 -> M=r, lanes 16-31 -> M=r+8; N=lane&15.
    #pragma unroll
    for (int mi = 0; mi < 2; ++mi) {
        #pragma unroll
        for (int ni = 0; ni < 4; ++ni) {
            int col = n0 + wn * 64 + ni * 16 + (lane & 15);
            float bv = bias ? bias[col] : 0.0f;
            #pragma unroll
            for (int r = 0; r < 8; ++r) {
                int row = m0 + wm * 32 + mi * 16 + r + 8 * (lane >> 4);
                long long idx = (long long)row * ldc + col;
                float v = acc[mi][ni][r] + bv;
                if (accum) v += C[idx];
                C[idx] = v;
            }
        }
    }
}

// ---------------------------------------------------- elementwise kernels --
__global__ void k_cvt_bf16(const float* __restrict__ s, bf16* __restrict__ d,
                           long long n) {
    long long i = blockIdx.x * 256LL + threadIdx.x;
    long long st = (long long)gridDim.x * 256LL;
    for (; i < n; i += st) d[i] = (bf16)s[i];
}

__global__ void k_transpose_bf16(const float* __restrict__ s, bf16* __restrict__ d,
                                 int R, int C) {
    long long n = (long long)R * C;
    long long i = blockIdx.x * 256LL + threadIdx.x;
    long long st = (long long)gridDim.x * 256LL;
    for (; i < n; i += st) {
        int r = (int)(i / C), c = (int)(i % C);
        d[(long long)c * R + r] = (bf16)s[i];
    }
}

// mid = silu(h[:, :F]) * h[:, F:2F]
__global__ void k_swiglu_mid(const float* __restrict__ h, bf16* __restrict__ mid,
                             long long n, int F) {
    long long i = blockIdx.x * 256LL + threadIdx.x;
    long long st = (long long)gridDim.x * 256LL;
    for (; i < n; i += st) {
        long long row = i / F; int col = (int)(i % F);
        float a = h[row * 2 * F + col];
        float b = h[row * 2 * F + F + col];
        float sg = 1.0f / (1.0f + __expf(-a));
        mid[i] = (bf16)(a * sg * b);
    }
}

// da = dmid*b*silu'(a), db = dmid*silu(a); optional f32 and bf16 outputs
__global__ void k_swiglu_bwd(const float* __restrict__ dmid,
                             const float* __restrict__ h,
                             float* __restrict__ df, bf16* __restrict__ dbf,
                             long long n, int F) {
    long long i = blockIdx.x * 256LL + threadIdx.x;
    long long st = (long long)gridDim.x * 256LL;
    for (; i < n; i += st) {
        long long row = i / F; int col = (int)(i % F);
        long long ia = row * 2 * F + col, ib = ia + F;
        float a = h[ia], b = h[ib], dm = dmid[i];
        float sg = 1.0f / (1.0f + __expf(-a));
        float da = dm * b * sg * (1.0f + a * (1.0f - sg));
        float db = dm * a * sg;
        if (df)  { df[ia] = da;        df[ib] = db; }
        if (dbf) { dbf[ia] = (bf16)da; dbf[ib] = (bf16)db; }
    }
}

// grad of mean(2+2*cos(a2,b2)) over T rows, wrt p rows; one block per row.
__global__ void k_loss_grad(const float* __restrict__ p, bf16* __restrict__ dp) {
    __shared__ float sd[256], sa[256], sb[256];
    int row = blockIdx.x, t = threadIdx.x;
    const float* pr = p + (long long)row * D_;
    float a = pr[t], b = pr[256 + t];
    sd[t] = a * b; sa[t] = a * a; sb[t] = b * b;
    __syncthreads();
    for (int s = 128; s > 0; s >>= 1) {
        if (t < s) { sd[t] += sd[t+s]; sa[t] += sa[t+s]; sb[t] += sb[t+s]; }
        __syncthreads();
    }
    float dot = sd[0];
    float Na = fmaxf(sqrtf(sa[0]), 1e-8f);
    float Nb = fmaxf(sqrtf(sb[0]), 1e-8f);
    float cosv = dot / (Na * Nb);
    float sc = 2.0f / (float)T_;              // d(mean(2+2cos))/dcos per row
    float ga = sc * (b / (Na * Nb) - cosv * a / (Na * Na));
    float gb = sc * (a / (Na * Nb) - cosv * b / (Nb * Nb));
    bf16* dr = dp + (long long)row * D_;
    dr[t] = (bf16)ga; dr[256 + t] = (bf16)gb;
}

// per-sample column sum over the two 'a' row-blocks (bias gradients)
__global__ void k_colsum(const float* __restrict__ src, float* __restrict__ out,
                         int C) {
    int col = blockIdx.x * 256 + threadIdx.x;
    int b = blockIdx.y;
    float s = 0.0f;
    for (int a = 0; a < 2; ++a) {
        const float* base = src + ((long long)(a * B_ + b) * T_) * C + col;
        for (int t = 0; t < T_; ++t) s += base[(long long)t * C];
    }
    out[(long long)b * C + col] = s;
}

__global__ void k_update_w(const float* __restrict__ u, const float* __restrict__ g,
                           const float* __restrict__ lr, bf16* __restrict__ f,
                           long long per, long long total) {
    float la = fabsf(lr[0]);
    long long i = blockIdx.x * 256LL + threadIdx.x;
    long long st = (long long)gridDim.x * 256LL;
    for (; i < total; i += st) f[i] = (bf16)(u[i % per] - la * g[i]);
}

__global__ void k_update_b(const float* __restrict__ u, const float* __restrict__ g,
                           const float* __restrict__ lr, float* __restrict__ f,
                           long long per, long long total) {
    float la = fabsf(lr[0]);
    long long i = blockIdx.x * 256LL + threadIdx.x;
    long long st = (long long)gridDim.x * 256LL;
    for (; i < total; i += st) f[i] = u[i % per] - la * g[i];
}

__global__ void k_layernorm(const float* __restrict__ x,
                            const float* __restrict__ gamma,
                            const float* __restrict__ beta,
                            float* __restrict__ out) {
    __shared__ float s1[256], s2[256];
    long long row = blockIdx.x; int t = threadIdx.x;
    const float* xr = x + row * D_;
    float v0 = xr[t], v1 = xr[t + 256];
    s1[t] = v0 + v1; s2[t] = v0 * v0 + v1 * v1;
    __syncthreads();
    for (int s = 128; s > 0; s >>= 1) {
        if (t < s) { s1[t] += s1[t+s]; s2[t] += s2[t+s]; }
        __syncthreads();
    }
    float mu  = s1[0] * (1.0f / D_);
    float var = s2[0] * (1.0f / D_) - mu * mu;
    float inv = rsqrtf(var + 1e-5f);
    float* o = out + row * D_;
    o[t]       = (v0 - mu) * inv * gamma[t]       + beta[t];
    o[t + 256] = (v1 - mu) * inv * gamma[t + 256] + beta[t + 256];
}

// ---------------------------------------------------------------- launch --
extern "C" void kernel_launch(void* const* d_in, const int* in_sizes, int n_in,
                              void* d_out, int out_size, void* d_ws, size_t ws_size,
                              hipStream_t stream) {
    (void)in_sizes; (void)n_in; (void)out_size; (void)ws_size;
    const float* x    = (const float*)d_in[0];
    const float* W1   = (const float*)d_in[1];
    const float* b1   = (const float*)d_in[2];
    const float* W2   = (const float*)d_in[3];
    const float* b2   = (const float*)d_in[4];
    const float* L    = (const float*)d_in[5];
    const float* Wp1  = (const float*)d_in[6];
    const float* bp1  = (const float*)d_in[7];
    const float* Wp2  = (const float*)d_in[8];
    const float* bp2  = (const float*)d_in[9];
    const float* uW1  = (const float*)d_in[10];
    const float* ub1  = (const float*)d_in[11];
    const float* uW2  = (const float*)d_in[12];
    const float* ub2  = (const float*)d_in[13];
    const float* lr   = (const float*)d_in[14];
    const float* gam  = (const float*)d_in[15];
    const float* bet  = (const float*)d_in[16];
    float* out = (float*)d_out;

    char* base = (char*)d_ws; size_t off = 0;
    auto alloc = [&](size_t bytes) -> char* {
        char* p = base + off; off += (bytes + 255) & ~(size_t)255; return p;
    };
    const long long NX = 32LL * T_ * D_;         // 16,777,216 (all x elems)
    bf16*  xbf   = (bf16*)alloc(NX * 2);
    bf16*  W1b   = (bf16*)alloc(1048576ULL * 2);
    bf16*  W2b   = (bf16*)alloc(524288ULL * 2);
    bf16*  Lb    = (bf16*)alloc(524288ULL * 2);
    bf16*  Wp1b  = (bf16*)alloc(524288ULL * 2);
    bf16*  Wp2b  = (bf16*)alloc(262144ULL * 2);
    bf16*  Wp2T  = (bf16*)alloc(262144ULL * 2);
    bf16*  Wp1T  = (bf16*)alloc(524288ULL * 2);
    bf16*  W2T   = (bf16*)alloc(524288ULL * 2);
    bf16*  LT    = (bf16*)alloc(524288ULL * 2);
    float* H1    = (float*)alloc(67108864ULL * 4);   // 32768 x 2048
    bf16*  Amid  = (bf16*)alloc(33554432ULL * 2);    // 32768 x 1024
    float* H     = (float*)alloc(16777216ULL * 4);   // 32768 x 512
    bf16*  Hbf   = (bf16*)alloc(16777216ULL * 2);
    float* yb    = (float*)alloc(8388608ULL * 4);    // 16384 x 512
    bf16*  ybf   = (bf16*)alloc(8388608ULL * 2);
    float* P1    = (float*)alloc(16777216ULL * 4);   // 16384 x 1024
    bf16*  Pmid  = (bf16*)alloc(8388608ULL * 2);
    float* pp    = (float*)alloc(8388608ULL * 4);
    bf16*  dpbf  = (bf16*)alloc(8388608ULL * 2);
    float* dPm   = (float*)alloc(8388608ULL * 4);
    bf16*  dP1b  = (bf16*)alloc(16777216ULL * 2);
    float* dy    = (float*)alloc(8388608ULL * 4);
    bf16*  dybf  = (bf16*)alloc(8388608ULL * 2);
    float* dH    = (float*)alloc(16777216ULL * 4);
    bf16*  dHbf  = (bf16*)alloc(16777216ULL * 2);
    float* g2    = (float*)alloc(8388608ULL * 4);    // 16 x 1024 x 512
    float* gb2   = (float*)alloc(8192ULL * 4);
    float* dAm   = (float*)alloc(33554432ULL * 4);
    float* dH1   = (float*)alloc(67108864ULL * 4);
    bf16*  dH1b  = (bf16*)alloc(67108864ULL * 2);
    float* g1    = (float*)alloc(16777216ULL * 4);   // 16 x 512 x 2048
    float* gb1   = (float*)alloc(32768ULL * 4);
    bf16*  fW1   = (bf16*)alloc(16777216ULL * 2);
    float* fb1   = (float*)alloc(32768ULL * 4);
    bf16*  fW2   = (bf16*)alloc(8388608ULL * 2);
    float* fb2   = (float*)alloc(8192ULL * 4);
    float* V1    = H1;     // reuse (H1 dead after dH1)
    bf16*  Vmid  = Amid;   // reuse (Amid dead after g2)
    float* Out   = H;      // reuse (H dead after ybf)

    auto g1d = [](long long n) {
        long long gb = (n + 255) / 256; if (gb > 8192) gb = 8192; return dim3((unsigned)gb);
    };
    auto GEMM = [&](const bf16* A, long long sA, int lda, int tA,
                    const bf16* Bm, long long sB, int ldb,
                    const float* bias, long long sBias,
                    float* C, long long sC, int ldc,
                    int M, int N, int K, int batch, int accum) {
        dim3 grid(N / BN, M / BM, batch);
        gemm_bf16<<<grid, 256, 0, stream>>>(A, sA, lda, tA, Bm, sB, ldb,
                                            bias, sBias, C, sC, ldc, M, N, K, accum);
    };
    const long long TS = (long long)T_;

    // ---- prep: bf16 copies + weight transposes ----
    k_cvt_bf16<<<g1d(NX), 256, 0, stream>>>(x, xbf, NX);
    k_cvt_bf16<<<g1d(1048576), 256, 0, stream>>>(W1, W1b, 1048576);
    k_cvt_bf16<<<g1d(524288), 256, 0, stream>>>(W2, W2b, 524288);
    k_cvt_bf16<<<g1d(524288), 256, 0, stream>>>(L, Lb, 524288);
    k_cvt_bf16<<<g1d(524288), 256, 0, stream>>>(Wp1, Wp1b, 524288);
    k_cvt_bf16<<<g1d(262144), 256, 0, stream>>>(Wp2, Wp2b, 262144);
    k_transpose_bf16<<<g1d(262144), 256, 0, stream>>>(Wp2, Wp2T, 512, 512);
    k_transpose_bf16<<<g1d(524288), 256, 0, stream>>>(Wp1, Wp1T, 512, 1024);
    k_transpose_bf16<<<g1d(524288), 256, 0, stream>>>(W2, W2T, 1024, 512);
    k_transpose_bf16<<<g1d(262144), 256, 0, stream>>>(L,          LT,          512, 512);
    k_transpose_bf16<<<g1d(262144), 256, 0, stream>>>(L + 262144, LT + 262144, 512, 512);

    // ---- forward inner loss path ----
    GEMM(xbf, 0, 512, 0, W1b, 0, 2048, b1, 0, H1, 0, 2048, 32768, 2048, 512, 1, 0);
    k_swiglu_mid<<<g1d(33554432), 256, 0, stream>>>(H1, Amid, 33554432LL, 1024);
    GEMM(Amid, 0, 1024, 0, W2b, 0, 512, b2, 0, H, 0, 512, 32768, 512, 1024, 1, 0);
    k_cvt_bf16<<<g1d(16777216), 256, 0, stream>>>(H, Hbf, 16777216);
    // y[b] = H[a=0,b]@L0 + H[a=1,b]@L1
    GEMM(Hbf,                 TS*512, 512, 0, Lb,          0, 512, nullptr, 0, yb, TS*512, 512, 1024, 512, 512, 16, 0);
    GEMM(Hbf + 16LL*TS*512,   TS*512, 512, 0, Lb + 262144, 0, 512, nullptr, 0, yb, TS*512, 512, 1024, 512, 512, 16, 1);
    k_cvt_bf16<<<g1d(8388608), 256, 0, stream>>>(yb, ybf, 8388608);
    GEMM(ybf, 0, 512, 0, Wp1b, 0, 1024, bp1, 0, P1, 0, 1024, 16384, 1024, 512, 1, 0);
    k_swiglu_mid<<<g1d(8388608), 256, 0, stream>>>(P1, Pmid, 8388608LL, 512);
    GEMM(Pmid, 0, 512, 0, Wp2b, 0, 512, bp2, 0, pp, 0, 512, 16384, 512, 512, 1, 0);

    // ---- backward to in_ff params ----
    k_loss_grad<<<16384, 256, 0, stream>>>(pp, dpbf);
    GEMM(dpbf, 0, 512, 0, Wp2T, 0, 512, nullptr, 0, dPm, 0, 512, 16384, 512, 512, 1, 0);
    k_swiglu_bwd<<<g1d(8388608), 256, 0, stream>>>(dPm, P1, nullptr, dP1b, 8388608LL, 512);
    GEMM(dP1b, 0, 1024, 0, Wp1T, 0, 512, nullptr, 0, dy, 0, 512, 16384, 512, 1024, 1, 0);
    k_cvt_bf16<<<g1d(8388608), 256, 0, stream>>>(dy, dybf, 8388608);
    GEMM(dybf, TS*512, 512, 0, LT,          0, 512, nullptr, 0, dH,                TS*512, 512, 1024, 512, 512, 16, 0);
    GEMM(dybf, TS*512, 512, 0, LT + 262144, 0, 512, nullptr, 0, dH + 16LL*TS*512,  TS*512, 512, 1024, 512, 512, 16, 0);
    k_cvt_bf16<<<g1d(16777216), 256, 0, stream>>>(dH, dHbf, 16777216);
    GEMM(dHbf, 0, 512, 0, W2T, 0, 1024, nullptr, 0, dAm, 0, 1024, 32768, 1024, 512, 1, 0);
    k_swiglu_bwd<<<g1d(33554432), 256, 0, stream>>>(dAm, H1, dH1, dH1b, 33554432LL, 1024);
    // g2[b] = Amid[b]^T @ dH[b]  (accumulate over a blocks)
    GEMM(Amid,                 TS*1024, 1024, 1, dHbf,               TS*512, 512, nullptr, 0, g2, 524288, 512, 1024, 512, 1024, 16, 0);
    GEMM(Amid + 16LL*TS*1024,  TS*1024, 1024, 1, dHbf + 16LL*TS*512, TS*512, 512, nullptr, 0, g2, 524288, 512, 1024, 512, 1024, 16, 1);
    k_colsum<<<dim3(2, 16), 256, 0, stream>>>(dH, gb2, 512);
    // g1[b] = x[b]^T @ dH1[b]
    GEMM(xbf,                TS*512, 512, 1, dH1b,                 TS*2048, 2048, nullptr, 0, g1, 1048576, 2048, 512, 2048, 1024, 16, 0);
    GEMM(xbf + 16LL*TS*512,  TS*512, 512, 1, dH1b + 16LL*TS*2048,  TS*2048, 2048, nullptr, 0, g1, 1048576, 2048, 512, 2048, 1024, 16, 1);
    k_colsum<<<dim3(8, 16), 256, 0, stream>>>(dH1, gb1, 2048);

    // ---- fast weights ----
    k_update_w<<<g1d(16777216), 256, 0, stream>>>(uW1, g1, lr, fW1, 1048576LL, 16777216LL);
    k_update_w<<<g1d(8388608),  256, 0, stream>>>(uW2, g2, lr, fW2, 524288LL, 8388608LL);
    k_update_b<<<g1d(32768),    256, 0, stream>>>(ub1, gb1, lr, fb1, 2048LL, 32768LL);
    k_update_b<<<g1d(8192),     256, 0, stream>>>(ub2, gb2, lr, fb2, 512LL, 8192LL);

    // ---- second forward with per-sample fast weights ----
    GEMM(xbf,               TS*512, 512, 0, fW1, 1048576, 2048, fb1, 2048, V1,                TS*2048, 2048, 1024, 2048, 512, 16, 0);
    GEMM(xbf + 16LL*TS*512, TS*512, 512, 0, fW1, 1048576, 2048, fb1, 2048, V1 + 16LL*TS*2048, TS*2048, 2048, 1024, 2048, 512, 16, 0);
    k_swiglu_mid<<<g1d(33554432), 256, 0, stream>>>(V1, Vmid, 33554432LL, 1024);
    GEMM(Vmid,                TS*1024, 1024, 0, fW2, 524288, 512, fb2, 512, Out,               TS*512, 512, 1024, 512, 1024, 16, 0);
    GEMM(Vmid + 16LL*TS*1024, TS*1024, 1024, 0, fW2, 524288, 512, fb2, 512, Out + 16LL*TS*512, TS*512, 512, 1024, 512, 1024, 16, 0);

    // ---- layernorm ----
    k_layernorm<<<32768, 256, 0, stream>>>(Out, gam, bet, out);
}
